// GCNConv_56573309223902
// MI455X (gfx1250) — matrix operand order repaired and checked by reading, not compiled, not run on hardware
//
#include <hip/hip_runtime.h>

#define NNODES 500000
#define CIN 128
#define COUT 128

typedef __attribute__((ext_vector_type(16))) __bf16 v16bf;
typedef __attribute__((ext_vector_type(8)))  float  v8f;

// ---------------- zero init (d_out is poisoned 0xAA by harness) ----------------
__global__ void zero_f32(float* __restrict__ p, long n) {
  long i = (long)blockIdx.x * blockDim.x + threadIdx.x;
  long stride = (long)gridDim.x * blockDim.x;
  for (; i < n; i += stride) p[i] = 0.0f;
}

// ---------------- degree: deg[row[e]] += 1 ----------------
__global__ void degree_kernel(const int* __restrict__ ei, float* __restrict__ deg, int n) {
  int e = blockIdx.x * blockDim.x + threadIdx.x;
  if (e >= n) return;
  unsigned r = (unsigned)ei[e];
  if (r < (unsigned)n) atomicAdd(&deg[r], 1.0f);
}

// ---------------- norm[e] = g(deg[row[e]]) * g(deg[col[e]]) ----------------
__global__ void norm_kernel(const int* __restrict__ ei, const float* __restrict__ deg,
                            float* __restrict__ norm, int n) {
  int e = blockIdx.x * blockDim.x + threadIdx.x;
  if (e >= n) return;
  unsigned r = (unsigned)ei[e];
  unsigned c = (unsigned)ei[n + e];
  float dr = (r < (unsigned)n) ? deg[r] : 0.0f;
  float dc = (c < (unsigned)n) ? deg[c] : 0.0f;
  float gr = (dr > 0.0f) ? rsqrtf(dr) : 0.0f;
  float gc = (dc > 0.0f) ? rsqrtf(dc) : 0.0f;
  norm[e] = gr * gc;
}

// ---------------- pre-swizzle W[out,in] -> B fragments (bf16) ----------------
// Bpack layout: fragment (kb,nt) for lane l is 16 contiguous bf16 at
//   Bpack[(((kb*8+nt)*32)+l)*16 + j]
// B-matrix 32x16 16-bit layout: lane holds column n = (l&15);
//   K = kb*32 + (j>>3)*16 + (l>>4)*8 + (j&7)   (lanes 0-15: K 0..7,16..23; lanes 16-31: K 8..15,24..31)
__global__ void pack_B(const float* __restrict__ W, __bf16* __restrict__ Bp) {
  int p = blockIdx.x * blockDim.x + threadIdx.x; // 4*8*32*16 = 16384
  if (p >= 16384) return;
  int j  = p & 15;
  int l  = (p >> 4) & 31;
  int nt = (p >> 9) & 7;
  int kb = p >> 12;
  int ncol = nt * 16 + (l & 15);
  int k    = kb * 32 + (j >> 3) * 16 + (l >> 4) * 8 + (j & 7);
  Bp[p] = (__bf16)W[ncol * CIN + k];           // W is [COUT, CIN] row-major
}

// ---------------- fused GEMM + bias + norm scaling (WMMA bf16) ----------------
// One wave -> 16 rows x 128 cols. 4 kblocks x 8 ntiles = 32 wmma per wave.
__global__ __launch_bounds__(256) void gemm_kernel(
    const float* __restrict__ x, const __bf16* __restrict__ Bp,
    const float* __restrict__ bias, const float* __restrict__ norm,
    float* __restrict__ out) {
  const int ntiles = NNODES / 16;                       // 31250 (exact)
  int wave = (int)((blockIdx.x * blockDim.x + threadIdx.x) >> 5);
  if (wave >= ntiles) return;                           // wave-uniform exit
  int lane = threadIdx.x & 31;
  int sub  = lane & 15;
  int hi   = lane >> 4;
  long m0  = (long)wave * 16;

  // A fragments: 16x32 bf16 per kblock. Lane (hi,sub) holds row m0+sub,
  // K = kb*32 + {hi*8+0..7, 16+hi*8+0..7}  (two contiguous 32B f32 chunks).
  const float* xrow = x + (m0 + sub) * CIN;
  v16bf a[4];
#pragma unroll
  for (int kb = 0; kb < 4; ++kb) {
    const float* p0 = xrow + kb * 32 + hi * 8;
    const float* p1 = xrow + kb * 32 + 16 + hi * 8;
#pragma unroll
    for (int j = 0; j < 8; ++j) {
      a[kb][j]     = (__bf16)p0[j];
      a[kb][j + 8] = (__bf16)p1[j];
    }
  }

  v8f acc[8];
  const v8f vzero = {0.f, 0.f, 0.f, 0.f, 0.f, 0.f, 0.f, 0.f};
#pragma unroll
  for (int nt = 0; nt < 8; ++nt) acc[nt] = vzero;

#pragma unroll
  for (int kb = 0; kb < 4; ++kb) {
#pragma unroll
    for (int nt = 0; nt < 8; ++nt) {
      v16bf bf = *(const v16bf*)(Bp + ((size_t)((kb * 8 + nt) * 32 + lane)) * 16);
      acc[nt] = __builtin_amdgcn_wmma_f32_16x16x32_bf16(
          /*neg_a=*/false, a[kb], /*neg_b=*/false, bf,
          /*c_mod=*/(short)0, acc[nt], /*reuse_a=*/false, /*reuse_b=*/false);
    }
  }

  // Epilogue: C/D layout -> VGPR r holds M = hi*8 + r, column N = sub (+16*nt).
  float nrm[8];
#pragma unroll
  for (int r = 0; r < 8; ++r) nrm[r] = norm[m0 + hi * 8 + r];
#pragma unroll
  for (int nt = 0; nt < 8; ++nt) {
    int col = nt * 16 + sub;
    float bv = bias[col];
#pragma unroll
    for (int r = 0; r < 8; ++r) {
      long row = m0 + hi * 8 + r;
      out[row * COUT + col] = (acc[nt][r] + bv) * nrm[r];
    }
  }
}

// ---------------- scatter: agg[row[e]] += out[col[e]] ----------------
// One thread per (edge, 4-channel group): float4 gather + 4x global_atomic_add_f32
__global__ void scatter_kernel(const int* __restrict__ ei, const float* __restrict__ out_tmp,
                               float* __restrict__ agg, int n) {
  long t = (long)blockIdx.x * blockDim.x + threadIdx.x;
  if (t >= (long)n * 32) return;
  int e  = (int)(t >> 5);
  int cg = ((int)t & 31) * 4;
  unsigned r = (unsigned)ei[e];
  unsigned c = (unsigned)ei[n + e];
  if (r >= (unsigned)n || c >= (unsigned)n) return;
  float4 v = *(const float4*)(out_tmp + (long)c * COUT + cg);
  float* dst = agg + (long)r * COUT + cg;
  atomicAdd(dst + 0, v.x);
  atomicAdd(dst + 1, v.y);
  atomicAdd(dst + 2, v.z);
  atomicAdd(dst + 3, v.w);
}

extern "C" void kernel_launch(void* const* d_in, const int* in_sizes, int n_in,
                              void* d_out, int out_size, void* d_ws, size_t ws_size,
                              hipStream_t stream) {
  const float* x  = (const float*)d_in[0];   // [N, 128]
  const int*   ei = (const int*)d_in[1];     // [2, N]
  const float* W  = (const float*)d_in[2];   // [128, 128]
  const float* b  = (const float*)d_in[3];   // [128]
  float* agg = (float*)d_out;                // [N, 128]
  const int n = NNODES;

  char* ws = (char*)d_ws;
  float*  deg     = (float*)(ws);                               // n floats
  float*  norm    = (float*)(ws + (size_t)n * 4);               // n floats
  __bf16* Bp      = (__bf16*)(ws + (size_t)n * 8);              // 32 KB
  float*  out_tmp = (float*)(ws + (size_t)n * 8 + 65536);       // n*128 floats

  // zero accumulator output and degree
  zero_f32<<<2048, 256, 0, stream>>>(agg, (long)n * COUT);
  zero_f32<<<512, 256, 0, stream>>>(deg, n);

  pack_B<<<64, 256, 0, stream>>>(W, Bp);

  degree_kernel<<<(n + 255) / 256, 256, 0, stream>>>(ei, deg, n);
  norm_kernel<<<(n + 255) / 256, 256, 0, stream>>>(ei, deg, norm, n);

  // 31250 row tiles, 8 waves (256 threads) per block
  const int ntiles = n / 16;
  const int blocks = (ntiles + 7) / 8;
  gemm_kernel<<<blocks, 256, 0, stream>>>(x, Bp, b, norm, out_tmp);

  long sthreads = (long)n * 32;
  scatter_kernel<<<(int)((sthreads + 255) / 256), 256, 0, stream>>>(ei, out_tmp, agg, n);
}
